// MultiHeadAttention_9990093930969
// MI455X (gfx1250) — matrix operand order repaired and checked by reading, not compiled, not run on hardware
//
#include <hip/hip_runtime.h>
#include <cstdint>
#include <cstddef>

// ---------- types ----------
typedef float v8f  __attribute__((ext_vector_type(8)));
typedef __bf16 v16bf __attribute__((ext_vector_type(16)));

struct __align__(16) U4 { unsigned int x, y, z, w; };
struct __align__(16) F4 { float x, y, z, w; };

union Frag32B { U4 q[2]; v16bf v; };   // 32 bytes: one WMMA bf16 A/B fragment per lane

struct AStage { U4 q0, q1, q2, q3; };  // one staged A row: 32 bf16
struct BStage { U4 q0, q1; };          // staged B chunk: 16 bf16

__device__ __forceinline__ unsigned short f2bf(float f) {
  unsigned int u = __float_as_uint(f);
  u += 0x7FFFu + ((u >> 16) & 1u);     // round-to-nearest-even
  return (unsigned short)(u >> 16);
}

__device__ __forceinline__ unsigned int pack2(float lo, float hi) {
  return (unsigned int)f2bf(lo) | ((unsigned int)f2bf(hi) << 16);
}

__device__ __forceinline__ AStage stage_a_bf16(const unsigned short* p) {
  const U4* s = (const U4*)p;
  AStage r; r.q0 = s[0]; r.q1 = s[1]; r.q2 = s[2]; r.q3 = s[3];
  return r;
}

__device__ __forceinline__ AStage stage_a_f32(const float* p) {
  const F4* s = (const F4*)p;
  F4 f0 = s[0], f1 = s[1], f2 = s[2], f3 = s[3];
  F4 f4 = s[4], f5 = s[5], f6 = s[6], f7 = s[7];
  AStage r;
  r.q0.x = pack2(f0.x, f0.y); r.q0.y = pack2(f0.z, f0.w);
  r.q0.z = pack2(f1.x, f1.y); r.q0.w = pack2(f1.z, f1.w);
  r.q1.x = pack2(f2.x, f2.y); r.q1.y = pack2(f2.z, f2.w);
  r.q1.z = pack2(f3.x, f3.y); r.q1.w = pack2(f3.z, f3.w);
  r.q2.x = pack2(f4.x, f4.y); r.q2.y = pack2(f4.z, f4.w);
  r.q2.z = pack2(f5.x, f5.y); r.q2.w = pack2(f5.z, f5.w);
  r.q3.x = pack2(f6.x, f6.y); r.q3.y = pack2(f6.z, f6.w);
  r.q3.z = pack2(f7.x, f7.y); r.q3.w = pack2(f7.z, f7.w);
  return r;
}

__device__ __forceinline__ BStage stage_b(const unsigned short* p) {
  const U4* s = (const U4*)p;
  BStage r; r.q0 = s[0]; r.q1 = s[1];
  return r;
}

// epilogue helper: one 16x16 C fragment slice (8 rows held by this lane)
__device__ __forceinline__ void store_acc(v8f a, float scale, const float* bias,
                                          size_t n, size_t mBase, int g,
                                          float* Cf, size_t cF, size_t ldc,
                                          unsigned short* Cb, size_t cB, size_t ldcb) {
  const float bv = bias ? bias[n] : 0.0f;
#pragma unroll
  for (int i = 0; i < 8; ++i) {
    const size_t m = mBase + (size_t)(g * 8 + i);
    const float val = a[i] * scale + bv;
    if (Cf) Cf[cF + m * ldc + n] = val;
    if (Cb) Cb[cB + m * ldcb + n] = f2bf(val);
  }
}

// ---------- fp32 -> bf16 convert ----------
__global__ void f32_to_bf16_k(const float* __restrict__ in,
                              unsigned short* __restrict__ out, int n) {
  int i = blockIdx.x * 256 + threadIdx.x;
  if (i < n) out[i] = f2bf(in[i]);
}

// ---------- WMMA GEMM ----------
// C[m,n] = scale * sum_k A[m,k]*B[k,n] + bias[n]
// A: bf16 (A_F32=0) or fp32 converted while staging (A_F32=1), row-major, ld = lda.
// B: bf16. B_TRANS=0: B is (K x N) row-major.  B_TRANS=1: B is (N x K) row-major (A @ B^T).
// Batched over gridDim.z; z -> (zb, zh) = (z/hpb, z%hpb) with separate strides.
// Block: 128 threads = 4 wave32; tile 128(M) x 64(N); K-step 32; wave owns 32x64 (8 WMMAs).
// Register-pipelined staging: next tile's global loads issue before the WMMA chain.
template <int A_F32, int B_TRANS>
__global__ __launch_bounds__(128)
void gemm_wmma(const void* __restrict__ Ap, size_t lda, size_t sAb, size_t sAh,
               const unsigned short* __restrict__ Bp, size_t ldb, size_t sBb, size_t sBh,
               float* __restrict__ Cf, size_t ldc, size_t sCfb, size_t sCfh,
               unsigned short* __restrict__ Cb, size_t ldcb, size_t sCbb, size_t sCbh,
               const float* __restrict__ bias, float scale, int K, int hpb) {
  const int LD = 40;                       // padded LDS row stride (shorts): 80B, 16B aligned
  __shared__ __align__(16) unsigned short As[128 * 40];  // A tile: 128 rows x 32 K
  __shared__ __align__(16) unsigned short Bs[64 * 40];   // B tile transposed: Bs[n][k]

  const int tid = threadIdx.x;
  const int z  = blockIdx.z;
  const int zb = z / hpb, zh = z - zb * hpb;
  const size_t rowBase = (size_t)blockIdx.y * 128;
  const size_t colBase = (size_t)blockIdx.x * 64;

  const size_t aOff = (size_t)zb * sAb + (size_t)zh * sAh + rowBase * lda;
  const size_t bOff = (size_t)zb * sBb + (size_t)zh * sBh +
                      (B_TRANS ? colBase * ldb : colBase);

  const int w    = tid >> 5;       // wave 0..3 -> rows [32w, 32w+32)
  const int lane = tid & 31;
  const int l16  = lane & 15;
  const int g    = lane >> 4;      // lane-group per ISA fragment layout

  // staging responsibilities
  const int rA = tid;                                   // A: one row of 32 bf16 per thread
  const int nB = tid >> 1, ksB = (tid & 1) << 4;        // B_TRANS: (n, 16-K-chunk)
  const int kkB = tid >> 2, nsB = (tid & 3) << 4;       // !B_TRANS: (k, 16-n-chunk)

  // incrementally-advanced source pointers
  const float*          aPtrF = (const float*)Ap + aOff + (size_t)rA * lda;
  const unsigned short* aPtrH = (const unsigned short*)Ap + aOff + (size_t)rA * lda;
  const unsigned short* bPtr  = B_TRANS
      ? Bp + bOff + (size_t)nB * ldb + (size_t)ksB
      : Bp + bOff + (size_t)kkB * ldb + (size_t)nsB;
  const size_t bStep = B_TRANS ? (size_t)32 : (size_t)32 * ldb;

  AStage ast;
  if (A_F32) ast = stage_a_f32(aPtrF); else ast = stage_a_bf16(aPtrH);
  BStage bst = stage_b(bPtr);

  const v8f vz = {0.f, 0.f, 0.f, 0.f, 0.f, 0.f, 0.f, 0.f};
  v8f c00 = vz, c01 = vz, c02 = vz, c03 = vz;
  v8f c10 = vz, c11 = vz, c12 = vz, c13 = vz;

  for (int k0 = 0; k0 < K; k0 += 32) {
    // ---- commit staged tile to LDS
    {
      U4* d = (U4*)&As[rA * LD];
      d[0] = ast.q0; d[1] = ast.q1; d[2] = ast.q2; d[3] = ast.q3;
    }
    if (B_TRANS) {
      U4* d = (U4*)&Bs[nB * LD + ksB];
      d[0] = bst.q0; d[1] = bst.q1;
    } else {
      union { U4 q[2]; unsigned short u[16]; } t;
      t.q[0] = bst.q0; t.q[1] = bst.q1;
#pragma unroll
      for (int j = 0; j < 16; ++j) Bs[(nsB + j) * LD + kkB] = t.u[j];
    }
    __syncthreads();

    // ---- issue next tile's global loads; in flight during the WMMA chain
    if (k0 + 32 < K) {
      aPtrF += 32; aPtrH += 32; bPtr += bStep;
      if (A_F32) ast = stage_a_f32(aPtrF); else ast = stage_a_bf16(aPtrH);
      bst = stage_b(bPtr);
      if (k0 + 64 < K) {
        if (A_F32) __builtin_prefetch(aPtrF + 32, 0, 1);   // global_prefetch_b8
        else       __builtin_prefetch(aPtrH + 32, 0, 1);
      }
    }

    // ---- hoist all fragments, then back-to-back WMMAs
    // A layout: lane holds row m; K = [16g, 16g+16) contiguous (matches B)
    Frag32B af0, af1, bu0, bu1, bu2, bu3;
    const unsigned short* ar0 = &As[(w * 32 + l16) * LD];
    const unsigned short* ar1 = &As[(w * 32 + 16 + l16) * LD];
    af0.q[0] = *(const U4*)(ar0 + 16 * g);
    af0.q[1] = *(const U4*)(ar0 + 16 * g + 8);
    af1.q[0] = *(const U4*)(ar1 + 16 * g);
    af1.q[1] = *(const U4*)(ar1 + 16 * g + 8);
    // B layout: lane holds col n=l16; K = [16g, 16g+16) contiguous
    const unsigned short* br = &Bs[l16 * LD + 16 * g];
    bu0.q[0] = *(const U4*)br;                 bu0.q[1] = *(const U4*)(br + 8);
    bu1.q[0] = *(const U4*)(br + 16 * LD);     bu1.q[1] = *(const U4*)(br + 16 * LD + 8);
    bu2.q[0] = *(const U4*)(br + 32 * LD);     bu2.q[1] = *(const U4*)(br + 32 * LD + 8);
    bu3.q[0] = *(const U4*)(br + 48 * LD);     bu3.q[1] = *(const U4*)(br + 48 * LD + 8);

    c00 = __builtin_amdgcn_wmma_f32_16x16x32_bf16(false, af0.v, false, bu0.v, (short)0, c00, false, false);
    c10 = __builtin_amdgcn_wmma_f32_16x16x32_bf16(false, af1.v, false, bu0.v, (short)0, c10, false, false);
    c01 = __builtin_amdgcn_wmma_f32_16x16x32_bf16(false, af0.v, false, bu1.v, (short)0, c01, false, false);
    c11 = __builtin_amdgcn_wmma_f32_16x16x32_bf16(false, af1.v, false, bu1.v, (short)0, c11, false, false);
    c02 = __builtin_amdgcn_wmma_f32_16x16x32_bf16(false, af0.v, false, bu2.v, (short)0, c02, false, false);
    c12 = __builtin_amdgcn_wmma_f32_16x16x32_bf16(false, af1.v, false, bu2.v, (short)0, c12, false, false);
    c03 = __builtin_amdgcn_wmma_f32_16x16x32_bf16(false, af0.v, false, bu3.v, (short)0, c03, false, false);
    c13 = __builtin_amdgcn_wmma_f32_16x16x32_bf16(false, af1.v, false, bu3.v, (short)0, c13, false, false);
    __syncthreads();
  }

  // ---- epilogue: C layout: lane holds n; VGPR i holds m = i + 8g
  const size_t cF = (size_t)zb * sCfb + (size_t)zh * sCfh;
  const size_t cB = (size_t)zb * sCbb + (size_t)zh * sCbh;
  const size_t m0 = rowBase + (size_t)(w * 32);       // first A-fragment rows
  const size_t m1 = m0 + 16;                          // second A-fragment rows
  const size_t n0 = colBase + (size_t)l16;
  store_acc(c00, scale, bias, n0 +  0, m0, g, Cf, cF, ldc, Cb, cB, ldcb);
  store_acc(c01, scale, bias, n0 + 16, m0, g, Cf, cF, ldc, Cb, cB, ldcb);
  store_acc(c02, scale, bias, n0 + 32, m0, g, Cf, cF, ldc, Cb, cB, ldcb);
  store_acc(c03, scale, bias, n0 + 48, m0, g, Cf, cF, ldc, Cb, cB, ldcb);
  store_acc(c10, scale, bias, n0 +  0, m1, g, Cf, cF, ldc, Cb, cB, ldcb);
  store_acc(c11, scale, bias, n0 + 16, m1, g, Cf, cF, ldc, Cb, cB, ldcb);
  store_acc(c12, scale, bias, n0 + 32, m1, g, Cf, cF, ldc, Cb, cB, ldcb);
  store_acc(c13, scale, bias, n0 + 48, m1, g, Cf, cF, ldc, Cb, cB, ldcb);
}

// ---------- row softmax over 2048 elements, in place ----------
__global__ __launch_bounds__(256) void softmax2048(float* __restrict__ attn) {
  const int tid = threadIdx.x;
  float* p = attn + (size_t)blockIdx.x * 2048;
  float x[8];
  float m = -3.0e38f;
#pragma unroll
  for (int j = 0; j < 8; ++j) { x[j] = p[tid + j * 256]; m = fmaxf(m, x[j]); }
#pragma unroll
  for (int off = 16; off > 0; off >>= 1) m = fmaxf(m, __shfl_xor(m, off, 32));
  __shared__ float rmax[8];
  __shared__ float rsum[8];
  const int w = tid >> 5, l = tid & 31;
  if (l == 0) rmax[w] = m;
  __syncthreads();
  m = rmax[0];
#pragma unroll
  for (int j = 1; j < 8; ++j) m = fmaxf(m, rmax[j]);
  float s = 0.f;
#pragma unroll
  for (int j = 0; j < 8; ++j) { x[j] = __expf(x[j] - m); s += x[j]; }
#pragma unroll
  for (int off = 16; off > 0; off >>= 1) s += __shfl_xor(s, off, 32);
  if (l == 0) rsum[w] = s;
  __syncthreads();
  s = 0.f;
#pragma unroll
  for (int j = 0; j < 8; ++j) s += rsum[j];
  const float inv = 1.0f / s;
#pragma unroll
  for (int j = 0; j < 8; ++j) p[tid + j * 256] = x[j] * inv;
}

// ---------- launcher ----------
extern "C" void kernel_launch(void* const* d_in, const int* in_sizes, int n_in,
                              void* d_out, int out_size, void* d_ws, size_t ws_size,
                              hipStream_t stream) {
  (void)in_sizes; (void)n_in; (void)out_size; (void)ws_size;
  const float* q  = (const float*)d_in[0];
  const float* k  = (const float*)d_in[1];
  const float* v  = (const float*)d_in[2];
  const float* Wq = (const float*)d_in[3];
  const float* bq = (const float*)d_in[4];
  const float* Wk = (const float*)d_in[5];
  const float* bk = (const float*)d_in[6];
  const float* Wv = (const float*)d_in[7];
  const float* bv = (const float*)d_in[8];
  const float* Wc = (const float*)d_in[9];
  const float* bc = (const float*)d_in[10];

  const int    Bn = 2, S = 2048, D = 1024, H = 16;
  const size_t rows = (size_t)Bn * S;       // 4096
  const size_t SS   = (size_t)S * S;        // 4194304
  const size_t inN  = rows * D;             // 4194304 elems
  const size_t wN   = (size_t)D * D;        // 1048576 elems

  // workspace layout (bf16/ushort elems), total 64 MiB
  unsigned short* ws  = (unsigned short*)d_ws;
  unsigned short* qb  = ws;
  unsigned short* kb  = qb  + inN;
  unsigned short* vb  = kb  + inN;
  unsigned short* wqb = vb  + inN;
  unsigned short* wkb = wqb + wN;
  unsigned short* wvb = wkb + wN;
  unsigned short* wcb = wvb + wN;
  unsigned short* qp  = wcb + wN;   // projected Q, bf16, (4096 x 1024)
  unsigned short* kp  = qp  + inN;
  unsigned short* vp  = kp  + inN;
  unsigned short* ao  = vp  + inN;  // attn output (concat), bf16

  float* out  = (float*)d_out;      // (4096 x 1024)
  float* attn = out + inN;          // (2 x 16 x 2048 x 2048)

  // ---- fp32 -> bf16 conversions
  const unsigned cgI = (unsigned)((inN + 255) / 256);
  const unsigned cgW = (unsigned)((wN + 255) / 256);
  f32_to_bf16_k<<<cgI, 256, 0, stream>>>(q,  qb,  (int)inN);
  f32_to_bf16_k<<<cgI, 256, 0, stream>>>(k,  kb,  (int)inN);
  f32_to_bf16_k<<<cgI, 256, 0, stream>>>(v,  vb,  (int)inN);
  f32_to_bf16_k<<<cgW, 256, 0, stream>>>(Wq, wqb, (int)wN);
  f32_to_bf16_k<<<cgW, 256, 0, stream>>>(Wk, wkb, (int)wN);
  f32_to_bf16_k<<<cgW, 256, 0, stream>>>(Wv, wvb, (int)wN);
  f32_to_bf16_k<<<cgW, 256, 0, stream>>>(Wc, wcb, (int)wN);

  const dim3 blk(128);

  // ---- Q/K/V projections: (4096x1024) @ (1024x1024) + b -> bf16
  const dim3 gproj(D / 64, (unsigned)(rows / 128), 1);   // 16 x 32
  gemm_wmma<0, 0><<<gproj, blk, 0, stream>>>(
      qb, D, 0, 0, wqb, D, 0, 0,
      nullptr, 0, 0, 0, qp, D, 0, 0, bq, 1.0f, D, 1);
  gemm_wmma<0, 0><<<gproj, blk, 0, stream>>>(
      kb, D, 0, 0, wkb, D, 0, 0,
      nullptr, 0, 0, 0, kp, D, 0, 0, bk, 1.0f, D, 1);
  gemm_wmma<0, 0><<<gproj, blk, 0, stream>>>(
      vb, D, 0, 0, wvb, D, 0, 0,
      nullptr, 0, 0, 0, vp, D, 0, 0, bv, 1.0f, D, 1);

  // ---- logits = scale * Qh @ Kh^T  -> fp32 into attn region, batched over 32 (b,h)
  const dim3 glog(S / 64, S / 128, Bn * H);              // 32 x 16 x 32
  gemm_wmma<0, 1><<<glog, blk, 0, stream>>>(
      qp, D, (size_t)S * D, 64,
      kp, D, (size_t)S * D, 64,
      attn, S, (size_t)H * SS, SS,
      nullptr, 0, 0, 0,
      nullptr, 0.125f /* 1/sqrt(64) */, 64, H);

  // ---- softmax rows (B*H*S rows of 2048), in place -> final attn_wgts output
  softmax2048<<<(unsigned)(rows * H), 256, 0, stream>>>(attn);

  // ---- attn @ Vh -> bf16 concat buffer, batched over 32 (b,h)
  const dim3 gav(1, S / 128, Bn * H);
  gemm_wmma<1, 0><<<gav, blk, 0, stream>>>(
      attn, S, (size_t)H * SS, SS,
      vp, D, (size_t)S * D, 64,
      nullptr, 0, 0, 0,
      ao, D, (size_t)S * D, 64,
      nullptr, 1.0f, S, H);

  // ---- output projection: (4096x1024) @ Wc + bc -> fp32 d_out
  gemm_wmma<0, 0><<<gproj, blk, 0, stream>>>(
      ao, D, 0, 0, wcb, D, 0, 0,
      out, D, 0, 0, nullptr, 0, 0, 0, bc, 1.0f, D, 1);
}
